// GNN_12910671691922
// MI455X (gfx1250) — matrix operand order repaired and checked by reading, not compiled, not run on hardware
//
#include <hip/hip_runtime.h>
#include <hip/hip_bf16.h>
#include <math.h>

typedef __attribute__((ext_vector_type(16))) _Float16 v16h;
typedef __attribute__((ext_vector_type(8)))  float    v8f;

namespace {
constexpr int kN  = 50000;   // nodes
constexpr int kF  = 128;     // input features
constexpr int kC  = 64;      // channels
constexpr int kH  = 4;       // heads
constexpr int kE  = 500000;  // edges
constexpr int kG  = 32;      // graphs
constexpr int kL  = 4;       // layers
constexpr int kHC = kH * kC; // 256
constexpr float kEps   = 1e-5f;
constexpr float kSlope = 0.2f;
}

// order-preserving float<->uint mapping for atomicMax-based segment max
__device__ __forceinline__ unsigned f32_to_ord(float f) {
  unsigned u = __float_as_uint(f);
  return (u & 0x80000000u) ? ~u : (u | 0x80000000u);
}
__device__ __forceinline__ float ord_to_f32(unsigned u) {
  return (u & 0x80000000u) ? __uint_as_float(u & 0x7FFFFFFFu)
                           : __uint_as_float(~u);
}

// ---------------------------------------------------------------------------
// One-time repack of a weight matrix B[K,Nc] (f32, row-major) into f16 WMMA
// B-fragment order: halves[((ntile*nks + ks)*32 + lane)*16 + j], where
// j -> K = ks*32 + 16*(lane>=16) + j, column = ntile*16 + (lane&15).
// Each lane's 16 halves are 32B-contiguous -> single v16h load in the GEMM.
// ---------------------------------------------------------------------------
__global__ void gnn_pack_b(const float* __restrict__ B, _Float16* __restrict__ Bp,
                           int K, int Nc) {
  const int t = blockIdx.x * blockDim.x + threadIdx.x;
  const int nks = K >> 5;
  const int total = (Nc >> 4) * nks * 32;
  if (t >= total) return;
  const int lane  = t & 31;
  const int ks    = (t >> 5) % nks;
  const int ntile = (t >> 5) / nks;
  const int col   = ntile * 16 + (lane & 15);
  const int kbase = ks * 32 + 16 * (lane >> 4);
  _Float16* dst = Bp + (size_t)t * 16;
#pragma unroll
  for (int j = 0; j < 16; ++j)
    dst[j] = (_Float16)B[(size_t)(kbase + j) * Nc + col];
}

// ---------------------------------------------------------------------------
// WMMA GEMM: C[M,Nc] = A[M,K] @ B[K,Nc] (+bias). Compile-time K/Nc so the
// K-loop fully unrolls. One wave computes a 16x64 output strip (4 acc tiles,
// A fragment reused across 4 WMMAs). A loaded as 4x float4 per k-step
// (contiguous A layout: one 16-float K run per lane);
// B fragment is a single 32B v16h load from the packed buffer.
// M % 16 == 0 (M = 50000 = 3125*16) -> no guards, EXEC all-ones for WMMA.
// ---------------------------------------------------------------------------
template <int K, int Nc>
__global__ void __launch_bounds__((Nc / 64) * 32)
gnn_wmma_gemm_t(const float* __restrict__ A, const _Float16* __restrict__ Bp,
                const float* __restrict__ bias, float* __restrict__ C, int M) {
  constexpr int kNT = 4;        // 16-col tiles per wave
  constexpr int nks = K / 32;   // k-steps
  const int lane = threadIdx.x & 31;
  const int l16  = lane & 15;
  const int lh   = lane >> 4;
  const int mrow0  = blockIdx.x * 16;
  const int ntile0 = threadIdx.y * kNT;
  const float* Arow = A + (size_t)(mrow0 + l16) * K;

  v8f acc[kNT] = {};
#pragma unroll
  for (int ks = 0; ks < nks; ++ks) {
    // A 16x32 fragment: lane covers K run [base..base+15]
    const float* ap = Arow + ks * 32 + 16 * lh;
    const float4 f0 = *(const float4*)(ap + 0);
    const float4 f1 = *(const float4*)(ap + 4);
    const float4 f2 = *(const float4*)(ap + 8);
    const float4 f3 = *(const float4*)(ap + 12);
    v16h a;
    a[0]  = (_Float16)f0.x; a[1]  = (_Float16)f0.y;
    a[2]  = (_Float16)f0.z; a[3]  = (_Float16)f0.w;
    a[4]  = (_Float16)f1.x; a[5]  = (_Float16)f1.y;
    a[6]  = (_Float16)f1.z; a[7]  = (_Float16)f1.w;
    a[8]  = (_Float16)f2.x; a[9]  = (_Float16)f2.y;
    a[10] = (_Float16)f2.z; a[11] = (_Float16)f2.w;
    a[12] = (_Float16)f3.x; a[13] = (_Float16)f3.y;
    a[14] = (_Float16)f3.z; a[15] = (_Float16)f3.w;
#pragma unroll
    for (int t = 0; t < kNT; ++t) {
      const v16h b = *(const v16h*)(
          Bp + ((size_t)((ntile0 + t) * nks + ks) * 32 + lane) * 16);
      acc[t] = __builtin_amdgcn_wmma_f32_16x16x32_f16(false, a, false, b,
                                                      (short)0, acc[t],
                                                      false, false);
    }
  }
#pragma unroll
  for (int t = 0; t < kNT; ++t) {
    const int col = (ntile0 + t) * 16 + l16;
    const float bv = bias ? bias[col] : 0.0f;
#pragma unroll
    for (int v = 0; v < 8; ++v) {  // C/D layout: VGPR v -> row = v + 8*lh
      C[(size_t)(mrow0 + v + 8 * lh) * Nc + col] = acc[t][v] + bv;
    }
  }
  (void)M;
}

// ---------------------------------------------------------------------------
// Edge phase 1: score[e,h] = att[h,:] . leaky_relu(xl[src,h,:] + xr[dst,h,:])
// and running segment max per (dst, head).
// ---------------------------------------------------------------------------
__global__ void gnn_edge_score_max(const long long* __restrict__ src,
                                   const long long* __restrict__ dst,
                                   const float* __restrict__ xl,
                                   const float* __restrict__ xr,
                                   const float* __restrict__ att,
                                   float* __restrict__ score,
                                   unsigned* __restrict__ nmax) {
  const int t = blockIdx.x * blockDim.x + threadIdx.x;
  if (t >= kE * kH) return;
  const int e = t >> 2, h = t & 3;
  const int s = (int)src[e], d = (int)dst[e];
  const float4* pl = (const float4*)(xl + (size_t)s * kHC + h * kC);
  const float4* pr = (const float4*)(xr + (size_t)d * kHC + h * kC);
  const float4* pa = (const float4*)(att + h * kC);
  __builtin_prefetch(pl, 0, 0);   // -> global_prefetch_b8
  float acc = 0.0f;
#pragma unroll
  for (int i = 0; i < kC / 4; ++i) {
    const float4 a = pl[i], b = pr[i], w = pa[i];
    float v;
    v = a.x + b.x; acc += w.x * (v > 0.f ? v : kSlope * v);
    v = a.y + b.y; acc += w.y * (v > 0.f ? v : kSlope * v);
    v = a.z + b.z; acc += w.z * (v > 0.f ? v : kSlope * v);
    v = a.w + b.w; acc += w.w * (v > 0.f ? v : kSlope * v);
  }
  score[t] = acc;
  atomicMax(&nmax[d * kH + h], f32_to_ord(acc));
}

// Edge phase 2: p = exp(score - max[dst,h]); accumulate softmax denominator.
__global__ void gnn_edge_exp_sum(const long long* __restrict__ dst,
                                 float* __restrict__ score,
                                 const unsigned* __restrict__ nmax,
                                 float* __restrict__ nsum) {
  const int t = blockIdx.x * blockDim.x + threadIdx.x;
  if (t >= kE * kH) return;
  const int e = t >> 2, h = t & 3;
  const int d = (int)dst[e];
  const float m = ord_to_f32(nmax[d * kH + h]);
  const float p = __expf(score[t] - m);
  score[t] = p;                       // reuse buffer as softmax numerator
  atomicAdd(&nsum[d * kH + h], p);
}

// Edge phase 3: hnext[dst,c] += (1/H) * sum_h alpha[e,h]*xl[src,h,c]
// (head-mean folded into the scatter -> 1 atomic per (edge, c)).
__global__ void gnn_edge_aggregate(const long long* __restrict__ src,
                                   const long long* __restrict__ dst,
                                   const float* __restrict__ xl,
                                   const float* __restrict__ p,
                                   const float* __restrict__ nsum,
                                   float* __restrict__ hnext) {
  const int t = blockIdx.x * blockDim.x + threadIdx.x;
  if (t >= kE * kC) return;
  const int e = t >> 6, c = t & (kC - 1);
  const int s = (int)src[e], d = (int)dst[e];
  float acc = 0.0f;
#pragma unroll
  for (int h = 0; h < kH; ++h) {
    const float alpha = p[e * kH + h] / nsum[d * kH + h];
    acc += alpha * xl[(size_t)s * kHC + h * kC + c];
  }
  atomicAdd(&hnext[(size_t)d * kC + c], 0.25f * acc);
}

// Graph-norm partial sums of (v + cb[c]) and its square over all N*C elems.
__global__ void gnn_norm_stats(const float* __restrict__ v,
                               const float* __restrict__ cb,
                               float* __restrict__ stats) {
  __shared__ float ss[256], ss2[256];
  float s = 0.f, s2 = 0.f;
  for (int i = blockIdx.x * blockDim.x + threadIdx.x; i < kN * kC;
       i += gridDim.x * blockDim.x) {
    const float val = v[i] + cb[i & (kC - 1)];
    s += val; s2 += val * val;
  }
  ss[threadIdx.x] = s; ss2[threadIdx.x] = s2;
  __syncthreads();
  for (int o = 128; o > 0; o >>= 1) {
    if ((int)threadIdx.x < o) {
      ss[threadIdx.x]  += ss[threadIdx.x + o];
      ss2[threadIdx.x] += ss2[threadIdx.x + o];
    }
    __syncthreads();
  }
  if (threadIdx.x == 0) {
    atomicAdd(&stats[0], ss[0]);
    atomicAdd(&stats[1], ss2[0]);
  }
}

__global__ void gnn_norm_finalize(float* __restrict__ stats) {
  const float cnt  = (float)kN * (float)kC;
  const float mean = stats[0] / cnt;
  float var = stats[1] / cnt - mean * mean;
  var = fmaxf(var, 0.f);
  stats[2] = mean;
  stats[3] = 1.0f / (sqrtf(var) + kEps);
}

// Fused graph-norm affine + SELU, writes h for next layer.
__global__ void gnn_norm_selu(const float* __restrict__ v,
                              const float* __restrict__ cb,
                              const float* __restrict__ lnw,
                              const float* __restrict__ lnb,
                              const float* __restrict__ stats,
                              float* __restrict__ h) {
  const int i = blockIdx.x * blockDim.x + threadIdx.x;
  if (i >= kN * kC) return;
  const int c = i & (kC - 1);
  const float y = (v[i] + cb[c] - stats[2]) * stats[3] * lnw[c] + lnb[c];
  const float SC = 1.0507009873554805f, AL = 1.6732632423543772f;
  h[i] = (y > 0.f) ? SC * y : SC * AL * (__expf(y) - 1.0f);
}

// Global mean pool accumulation.
__global__ void gnn_pool_accum(const float* __restrict__ h,
                               const long long* __restrict__ batch,
                               float* __restrict__ pooled,
                               float* __restrict__ counts) {
  const int i = blockIdx.x * blockDim.x + threadIdx.x;
  if (i >= kN * kC) return;
  const int n = i >> 6, c = i & (kC - 1);
  const int g = (int)batch[n];
  atomicAdd(&pooled[g * kC + c], h[i]);
  if (c == 0) atomicAdd(&counts[g], 1.0f);
}

__global__ void gnn_pool_head(const float* __restrict__ pooled,
                              const float* __restrict__ counts,
                              const float* __restrict__ Wh,
                              const float* __restrict__ bh,
                              float* __restrict__ out) {
  const int g = threadIdx.x;
  if (g >= kG) return;
  const float cnt = fmaxf(counts[g], 1.0f);
  float acc = 0.0f;
  for (int c = 0; c < kC; ++c) acc += (pooled[g * kC + c] / cnt) * Wh[c];
  out[g] = acc + bh[0];
}

// ---------------------------------------------------------------------------
extern "C" void kernel_launch(void* const* d_in, const int* in_sizes, int n_in,
                              void* d_out, int out_size, void* d_ws,
                              size_t ws_size, hipStream_t stream) {
  (void)in_sizes; (void)n_in; (void)out_size; (void)ws_size;
  const float*     x     = (const float*)d_in[0];
  const long long* ei    = (const long long*)d_in[1];   // [2, E] int64
  const long long* batch = (const long long*)d_in[2];
  const float* Wp  = (const float*)d_in[3];
  const float* Wl  = (const float*)d_in[4];
  const float* bl  = (const float*)d_in[5];
  const float* Wr  = (const float*)d_in[6];
  const float* br  = (const float*)d_in[7];
  const float* att = (const float*)d_in[8];
  const float* cb  = (const float*)d_in[9];
  const float* lnw = (const float*)d_in[10];
  const float* lnb = (const float*)d_in[11];
  const float* Wh  = (const float*)d_in[12];
  const float* bh  = (const float*)d_in[13];
  float* out = (float*)d_out;
  const long long* srcIdx = ei;
  const long long* dstIdx = ei + kE;

  // workspace carving (256B aligned); total ~138 MB -> L2-resident hot set
  char* ws = (char*)d_ws;
  size_t off = 0;
  auto take = [&](size_t bytes) -> void* {
    void* p = ws + off;
    off = (off + bytes + 255) & ~(size_t)255;
    return p;
  };
  float*     h      = (float*)take((size_t)kN * kC * 4);
  float*     hnext  = (float*)take((size_t)kN * kC * 4);
  float*     xl     = (float*)take((size_t)kN * kHC * 4);
  float*     xr     = (float*)take((size_t)kN * kHC * 4);
  float*     score  = (float*)take((size_t)kE * kH * 4);
  unsigned*  nmax   = (unsigned*)take((size_t)kN * kH * 4);
  float*     nsum   = (float*)take((size_t)kN * kH * 4);
  float*     stats  = (float*)take(4 * sizeof(float));
  float*     pooled = (float*)take((size_t)kG * kC * 4);
  float*     counts = (float*)take((size_t)kG * 4);
  _Float16*  packP  = (_Float16*)take((size_t)(kC / 16) * (kF / 32) * 32 * 16 * 2);
  _Float16*  packL  = (_Float16*)take((size_t)(kHC / 16) * (kC / 32) * 32 * 16 * 2);
  _Float16*  packR  = (_Float16*)take((size_t)(kHC / 16) * (kC / 32) * 32 * 16 * 2);

  const int EH_blocks = (kE * kH + 255) / 256;
  const int EC_blocks = (kE * kC + 255) / 256;
  const int NC_blocks = (kN * kC + 255) / 256;

  // initial projection: h = x @ Wp  (no bias); Wp packed to WMMA fragments
  {
    const int total = (kC / 16) * (kF / 32) * 32;   // 4*4*32 = 512 threads
    gnn_pack_b<<<(total + 255) / 256, 256, 0, stream>>>(Wp, packP, kF, kC);
    gnn_wmma_gemm_t<kF, kC><<<dim3(kN / 16), dim3(32, 1), 0, stream>>>(
        x, packP, nullptr, h, kN);
  }

  for (int l = 0; l < kL; ++l) {
    hipMemsetAsync(nmax,  0, (size_t)kN * kH * 4, stream);
    hipMemsetAsync(nsum,  0, (size_t)kN * kH * 4, stream);
    hipMemsetAsync(hnext, 0, (size_t)kN * kC * 4, stream);
    hipMemsetAsync(stats, 0, 4 * sizeof(float), stream);

    const float* Wl_l  = Wl  + (size_t)l * kC * kHC;
    const float* bl_l  = bl  + (size_t)l * kHC;
    const float* Wr_l  = Wr  + (size_t)l * kC * kHC;
    const float* br_l  = br  + (size_t)l * kHC;
    const float* att_l = att + (size_t)l * kH * kC;
    const float* cb_l  = cb  + (size_t)l * kC;
    const float* lnw_l = lnw + (size_t)l * kC;
    const float* lnb_l = lnb + (size_t)l * kC;

    // pack this layer's weights, then per-head projections [N,64]@[64,256]+b
    const int totalLR = (kHC / 16) * (kC / 32) * 32;   // 16*2*32 = 1024
    gnn_pack_b<<<(totalLR + 255) / 256, 256, 0, stream>>>(Wl_l, packL, kC, kHC);
    gnn_pack_b<<<(totalLR + 255) / 256, 256, 0, stream>>>(Wr_l, packR, kC, kHC);
    gnn_wmma_gemm_t<kC, kHC><<<dim3(kN / 16), dim3(32, 4), 0, stream>>>(
        h, packL, bl_l, xl, kN);
    gnn_wmma_gemm_t<kC, kHC><<<dim3(kN / 16), dim3(32, 4), 0, stream>>>(
        h, packR, br_l, xr, kN);

    gnn_edge_score_max<<<EH_blocks, 256, 0, stream>>>(
        srcIdx, dstIdx, xl, xr, att_l, score, nmax);
    gnn_edge_exp_sum<<<EH_blocks, 256, 0, stream>>>(dstIdx, score, nmax, nsum);
    gnn_edge_aggregate<<<EC_blocks, 256, 0, stream>>>(
        srcIdx, dstIdx, xl, score, nsum, hnext);

    gnn_norm_stats<<<1024, 256, 0, stream>>>(hnext, cb_l, stats);
    gnn_norm_finalize<<<1, 1, 0, stream>>>(stats);
    gnn_norm_selu<<<NC_blocks, 256, 0, stream>>>(
        hnext, cb_l, lnw_l, lnb_l, stats, h);
  }

  hipMemsetAsync(pooled, 0, (size_t)kG * kC * 4, stream);
  hipMemsetAsync(counts, 0, (size_t)kG * 4, stream);
  gnn_pool_accum<<<NC_blocks, 256, 0, stream>>>(h, batch, pooled, counts);
  gnn_pool_head<<<1, kG, 0, stream>>>(pooled, counts, Wh, bh, out);
}
